// ReasoningModule_55946243998361
// MI455X (gfx1250) — compile-verified, matrix operand order checked by reading, not run on hardware
//
#include <hip/hip_runtime.h>
#include <hip/hip_bf16.h>
#include <cstdint>
#include <cstddef>

typedef __bf16 bf16;
typedef bf16 v16bf __attribute__((ext_vector_type(16)));
typedef bf16 v8bf  __attribute__((ext_vector_type(8)));
typedef float v8f  __attribute__((ext_vector_type(8)));
typedef float v4f  __attribute__((ext_vector_type(4)));
typedef int  vs4i  __attribute__((vector_size(16)));   // matches async builtin param type

#define AS1 __attribute__((address_space(1)))
#define AS3 __attribute__((address_space(3)))

// ---- gfx1250 feature guards (host pass falls back automatically) ----
#if defined(__AMDGCN__) && __has_builtin(__builtin_amdgcn_global_load_async_to_lds_b128)
#define HAVE_ASYNC_LDS 1
#else
#define HAVE_ASYNC_LDS 0
#endif
#if defined(__AMDGCN__) && __has_builtin(__builtin_amdgcn_s_wait_asynccnt)
#define WAIT_ASYNC() __builtin_amdgcn_s_wait_asynccnt(0)
#else
#define WAIT_ASYNC() do {} while (0)
#endif
#if defined(__AMDGCN__) && __has_builtin(__builtin_amdgcn_permlane16)
#define HAVE_PERMLANE16 1
#else
#define HAVE_PERMLANE16 0
#endif

namespace {

constexpr int Bb = 2, Ss = 2048, Dd = 1024, DI = 2048;
constexpr int Ee = 8, FF = 4096, NBv = 4, NH = 16, HD = 64, Ll = 1024;
constexpr int TOK  = Bb * Ss;   // 4096 tokens
constexpr int TOKH = Bb * Ll;   // 2048 half-rate tokens
constexpr float SKIPT = 0.3f;
constexpr float HAARI = 0.7071067811865476f;

// ---- fp32 workspace layout (element offsets) ----
constexpr size_t OFF_MASK = 0;
constexpr size_t OFF_POOL = 64;
constexpr size_t OFF_XLN  = 4096;
constexpr size_t SZ_XLN   = (size_t)TOK * Dd;
constexpr size_t OFF_BIG  = OFF_XLN + SZ_XLN;      // xz / tc out
constexpr size_t SZ_BIG   = (size_t)TOK * FF;
constexpr size_t OFF_XC   = OFF_BIG + SZ_BIG;      // xc / o2
constexpr size_t SZ_XC    = (size_t)TOK * DI;
constexpr size_t OFF_DT   = OFF_XC + SZ_XC;        // dt / det
constexpr size_t SZ_DT    = (size_t)TOK * DI;
constexpr size_t OFF_BC   = OFF_DT + SZ_DT;        // bc / logits+gates
constexpr size_t SZ_BC    = (size_t)TOK * 128;
constexpr size_t OFF_Y    = OFF_BC + SZ_BC;        // wlam y (f32)
constexpr size_t SZ_Y     = (size_t)TOK * DI;
constexpr size_t OFF_BLK  = OFF_Y + SZ_Y;          // branch output accum
constexpr size_t SZ_BLK   = (size_t)TOK * Dd;
// bf16 region starts here (addressed as bf16*)
constexpr size_t OFF_BFB  = OFF_BLK + SZ_BLK;
constexpr size_t BF_XLN   = 0;                     // TOK*Dd bf16
constexpr size_t BF_SCR   = (size_t)TOK * Dd;      // shared bf16 scratch, TOK*FF elems

// ---------- device helpers ----------
__device__ __forceinline__ float sigmoidf_(float x){ return 1.f/(1.f+__expf(-x)); }
__device__ __forceinline__ float geluf_(float x){
  float x3 = x*x*x;
  return 0.5f*x*(1.f + tanhf(0.7978845608028654f*(x + 0.044715f*x3)));
}
__device__ __forceinline__ float softplusf_(float x){
  return (x > 20.f) ? x : log1pf(__expf(x));
}

// async (or fallback) 16-byte global->LDS copy of bf16 data
__device__ __forceinline__ void cp_b128_g2l(const bf16* __restrict__ src, bf16* __restrict__ dst){
#if HAVE_ASYNC_LDS
  __builtin_amdgcn_global_load_async_to_lds_b128((AS1 vs4i*)src, (AS3 vs4i*)dst, 0, 0);
#else
  *(v8bf*)dst = *(const v8bf*)src;
#endif
}

// xor-butterfly reductions confined to 16-lane rows (wave32 halves)
#if HAVE_PERMLANE16
template<unsigned LO, unsigned HI>
__device__ __forceinline__ float pl16_(float v){
  unsigned u = __builtin_bit_cast(unsigned, v);
  unsigned r = __builtin_amdgcn_permlane16(u, u, LO, HI, false, false);
  return __builtin_bit_cast(float, r);
}
__device__ __forceinline__ float rowred_max(float v){
  v = fmaxf(v, pl16_<0x67452301u, 0xEFCDAB89u>(v));   // xor 1
  v = fmaxf(v, pl16_<0x45670123u, 0xCDEF89ABu>(v));   // xor 3 (same reduction as xor 2 after xor 1)
  v = fmaxf(v, pl16_<0x32107654u, 0xBA98FEDCu>(v));   // xor 4
  v = fmaxf(v, pl16_<0xFEDCBA98u, 0x76543210u>(v));   // xor 8
  return v;
}
__device__ __forceinline__ float rowred_sum(float v){
  v += pl16_<0x67452301u, 0xEFCDAB89u>(v);
  v += pl16_<0x45670123u, 0xCDEF89ABu>(v);
  v += pl16_<0x32107654u, 0xBA98FEDCu>(v);
  v += pl16_<0xFEDCBA98u, 0x76543210u>(v);
  return v;
}
#else
__device__ __forceinline__ float rowred_max(float v){
  v = fmaxf(v, __shfl_xor(v, 1)); v = fmaxf(v, __shfl_xor(v, 2));
  v = fmaxf(v, __shfl_xor(v, 4)); v = fmaxf(v, __shfl_xor(v, 8));
  return v;
}
__device__ __forceinline__ float rowred_sum(float v){
  v += __shfl_xor(v, 1); v += __shfl_xor(v, 2);
  v += __shfl_xor(v, 4); v += __shfl_xor(v, 8);
  return v;
}
#endif

// Load a 16x32 bf16 A/B fragment from row-major LDS (ldm = row stride in elems).
__device__ __forceinline__ v16bf frag_ld(const bf16* __restrict__ base, int ldm, int lane){
  int r = lane & 15, half = lane >> 4;
  const bf16* p = base + r*ldm + half*8;
  v8bf lo = *(const v8bf*)(p);
  v8bf hi = *(const v8bf*)(p + 16);
  return __builtin_shufflevector(lo, hi, 0,1,2,3,4,5,6,7,8,9,10,11,12,13,14,15);
}
#define WMMA_BF16(a,b,c) \
  __builtin_amdgcn_wmma_f32_16x16x32_bf16(false,(a),false,(b),(short)0,(c),false,false)

// ================== generic bf16 WMMA GEMM ==================
// C[M,N] = epi(A[M,K] @ B[K,N]); A is bf16, B is fp32 (converted in staging).
enum Epi { EPI_NONE = 0, EPI_BIAS = 1, EPI_GELU = 2, EPI_GATE_ACC = 3 };
constexpr int BM = 128, BN = 128, BK = 32;

template<int EPI, bool USE_GATE, bool OBF>
__global__ __launch_bounds__(256)
void gemm_bf16(const bf16* __restrict__ A, const float* __restrict__ Bm,
               const float* __restrict__ bias, const float* __restrict__ gate,
               int gate_stride, void* __restrict__ Cv, int M, int N, int K)
{
  __shared__ __attribute__((aligned(16))) bf16 sA[BM][BK];
  __shared__ __attribute__((aligned(16))) bf16 sB[BN][BK];   // stored N-major (B^T)
  __shared__ float sGate[BM];
  __shared__ int sAny;

  const int tid = threadIdx.x;
  const int bm = blockIdx.y * BM;
  const int bn = blockIdx.x * BN;

  if (USE_GATE) {
    if (tid == 0) sAny = 0;
    __syncthreads();
    if (tid < BM) {
      float g = gate[(size_t)(bm + tid) * gate_stride];
      sGate[tid] = g;
      if (g != 0.0f) atomicOr(&sAny, 1);
    }
    __syncthreads();
    if (!sAny) return;        // whole block gated off (MoE top-k skip)
  }

  const int lane = tid & 31;
  const int wave = tid >> 5;              // 8 waves: 4 in M x 2 in N
  const int wm = (wave >> 1) * 32;
  const int wn = (wave & 1) * 64;

  v8f acc[2][4];
  #pragma unroll
  for (int i = 0; i < 2; i++)
    #pragma unroll
    for (int j = 0; j < 4; j++)
      #pragma unroll
      for (int e = 0; e < 8; e++) acc[i][j][e] = 0.f;

  for (int k0 = 0; k0 < K; k0 += BK) {
    { // A tile (128x32 bf16 = 8KB): 512 x 16B chunks, async copy global->LDS
      #pragma unroll
      for (int ci = tid; ci < 512; ci += 256) {
        int r = ci >> 2, o = (ci & 3) * 8;
        cp_b128_g2l(A + (size_t)(bm + r) * K + k0 + o, &sA[r][o]);
      }
    }
    { // B tile (32x128 f32) transposed + converted -> sB[n][k] bf16
      int kk = tid >> 3, n0 = (tid & 7) * 16;
      const float* bp = Bm + (size_t)(k0 + kk) * N + bn + n0;
      if (k0 + BK < K) __builtin_prefetch(bp + (size_t)BK * N, 0, 1);
      #pragma unroll
      for (int i = 0; i < 16; i += 4) {
        v4f v = *(const v4f*)(bp + i);
        sB[n0+i+0][kk] = (bf16)v.x; sB[n0+i+1][kk] = (bf16)v.y;
        sB[n0+i+2][kk] = (bf16)v.z; sB[n0+i+3][kk] = (bf16)v.w;
      }
    }
    WAIT_ASYNC();
    __syncthreads();

    v16bf af[2], bfg[4];
    #pragma unroll
    for (int i = 0; i < 2; i++) af[i]  = frag_ld(&sA[wm + i*16][0], BK, lane);
    #pragma unroll
    for (int j = 0; j < 4; j++) bfg[j] = frag_ld(&sB[wn + j*16][0], BK, lane);
    #pragma unroll
    for (int i = 0; i < 2; i++)
      #pragma unroll
      for (int j = 0; j < 4; j++)
        acc[i][j] = WMMA_BF16(af[i], bfg[j], acc[i][j]);
    __syncthreads();
  }

  float* Cf = (float*)Cv;
  bf16*  Cb = (bf16*)Cv;
  const int col_lo = lane & 15, halfsel = lane >> 4;
  #pragma unroll
  for (int i = 0; i < 2; i++) {
    #pragma unroll
    for (int j = 0; j < 4; j++) {
      int col = bn + wn + j*16 + col_lo;
      #pragma unroll
      for (int e = 0; e < 8; e++) {
        int rloc = wm + i*16 + (halfsel ? e + 8 : e);
        int row  = bm + rloc;
        float v = acc[i][j][e];
        if (EPI != EPI_NONE) v += bias[col];
        if (EPI == EPI_GELU) v = geluf_(v);
        size_t idx = (size_t)row * N + col;
        if (EPI == EPI_GATE_ACC)      Cf[idx] += sGate[rloc] * v;
        else if (OBF)                 Cb[idx]  = (bf16)v;
        else                          Cf[idx]  = v;
      }
    }
  }
}

// ================== flash attention (WLAM), bf16 in/out ==================
__global__ __launch_bounds__(256)
void attn_kernel(const bf16* __restrict__ Q, const bf16* __restrict__ Kg,
                 const bf16* __restrict__ Vg, bf16* __restrict__ O)
{
  __shared__ __attribute__((aligned(16))) bf16 sQ[128][64];
  __shared__ __attribute__((aligned(16))) bf16 sK[32][64];
  __shared__ __attribute__((aligned(16))) bf16 sVt[64][32];   // v transposed
  __shared__ __attribute__((aligned(16))) bf16 sP[8][16][32]; // per-wave P tile

  const int tid = threadIdx.x, lane = tid & 31, wave = tid >> 5;
  const int qb = blockIdx.x;
  const int bh = blockIdx.y;
  const int b = bh / NH, h = bh % NH;
  const size_t rowbase = (size_t)b * Ll;
  const int cbase = h * HD;

  // stage Q (128x64 bf16) via async b128 copies
  #pragma unroll
  for (int ci = tid; ci < 1024; ci += 256) {
    int r = ci >> 3, o = (ci & 7) * 8;
    cp_b128_g2l(Q + (rowbase + qb*128 + r) * Dd + cbase + o, &sQ[r][o]);
  }
  WAIT_ASYNC();
  __syncthreads();

  v16bf qf0 = frag_ld(&sQ[wave*16][0],  64, lane);
  v16bf qf1 = frag_ld(&sQ[wave*16][32], 64, lane);

  float mrow[8], lrow[8];
  v8f accO[4];
  #pragma unroll
  for (int e = 0; e < 8; e++) { mrow[e] = -1e30f; lrow[e] = 0.f; }
  #pragma unroll
  for (int j = 0; j < 4; j++)
    #pragma unroll
    for (int e = 0; e < 8; e++) accO[j][e] = 0.f;

  for (int kb = 0; kb < Ll; kb += 32) {
    { // K tile: 32x64 bf16 = 256 x 16B chunks, one per thread (async)
      int r = tid >> 3, o = (tid & 7) * 8;
      cp_b128_g2l(Kg + (rowbase + kb + r) * Dd + cbase + o, &sK[r][o]);
    }
    for (int i = tid; i < 32*64; i += 256) { // V transposed (scalar stores)
      int r = i >> 6, c = i & 63;
      sVt[c][r] = Vg[(rowbase + kb + r) * Dd + cbase + c];
    }
    WAIT_ASYNC();
    __syncthreads();

    v8f accS[2];
    #pragma unroll
    for (int t = 0; t < 2; t++)
      #pragma unroll
      for (int e = 0; e < 8; e++) accS[t][e] = 0.f;
    #pragma unroll
    for (int t = 0; t < 2; t++) {
      v16bf kf0 = frag_ld(&sK[t*16][0],  64, lane);
      v16bf kf1 = frag_ld(&sK[t*16][32], 64, lane);
      accS[t] = WMMA_BF16(qf0, kf0, accS[t]);
      accS[t] = WMMA_BF16(qf1, kf1, accS[t]);
    }

    // online softmax; row reductions via v_permlane16 (stay within 16-lane half)
    float sc[8];
    #pragma unroll
    for (int e = 0; e < 8; e++) {
      float v0 = accS[0][e] * 0.125f, v1 = accS[1][e] * 0.125f;
      float mx = rowred_max(fmaxf(v0, v1));
      float mnew = fmaxf(mrow[e], mx);
      sc[e] = __expf(mrow[e] - mnew);
      float p0 = __expf(v0 - mnew), p1 = __expf(v1 - mnew);
      accS[0][e] = p0; accS[1][e] = p1;
      lrow[e] = lrow[e] * sc[e] + rowred_sum(p0 + p1);
      mrow[e] = mnew;
    }
    #pragma unroll
    for (int j = 0; j < 4; j++)
      #pragma unroll
      for (int e = 0; e < 8; e++) accO[j][e] *= sc[e];

    { // P (C-layout) -> per-wave LDS -> A-fragment layout
      int col = lane & 15, half = lane >> 4;
      #pragma unroll
      for (int e = 0; e < 8; e++) {
        int r = half ? e + 8 : e;
        sP[wave][r][col]      = (bf16)accS[0][e];
        sP[wave][r][16 + col] = (bf16)accS[1][e];
      }
    }
    v16bf pf = frag_ld(&sP[wave][0][0], 32, lane);
    #pragma unroll
    for (int j = 0; j < 4; j++) {
      v16bf vf = frag_ld(&sVt[j*16][0], 32, lane);
      accO[j] = WMMA_BF16(pf, vf, accO[j]);
    }
    __syncthreads();
  }

  const int col = lane & 15, half = lane >> 4;
  #pragma unroll
  for (int j = 0; j < 4; j++)
    #pragma unroll
    for (int e = 0; e < 8; e++) {
      int rloc = wave*16 + (half ? e + 8 : e);
      O[(rowbase + qb*128 + rloc) * Dd + cbase + j*16 + col] =
          (bf16)(accO[j][e] / lrow[e]);
    }
}

// ================== small / elementwise kernels ==================
__global__ void copy_kernel(const float* __restrict__ in, float* __restrict__ out, int n){
  int i = blockIdx.x*256 + threadIdx.x; if (i < n) out[i] = in[i];
}
__global__ void zero_kernel(float* __restrict__ p, int n){
  int i = blockIdx.x*256 + threadIdx.x; if (i < n) p[i] = 0.f;
}
__global__ void pool_kernel(const float* __restrict__ h, float* __restrict__ pooled){
  int i = blockIdx.x*256 + threadIdx.x;
  if (i >= Bb*Dd) return;
  int b = i / Dd, d = i % Dd;
  float acc = 0.f;
  for (int s = 0; s < Ss; s++) acc += h[((size_t)b*Ss + s)*Dd + d];
  pooled[i] = acc * (1.f/Ss);
}
__global__ void router_kernel(const float* __restrict__ pooled, const float* __restrict__ rw,
                              const float* __restrict__ rb, float* __restrict__ masks){
  __shared__ float red[256];
  int b = blockIdx.x / NBv, j = blockIdx.x % NBv;
  float acc = 0.f;
  for (int d = threadIdx.x; d < Dd; d += 256) acc += pooled[b*Dd + d] * rw[d*NBv + j];
  red[threadIdx.x] = acc; __syncthreads();
  for (int s = 128; s > 0; s >>= 1){
    if (threadIdx.x < s) red[threadIdx.x] += red[threadIdx.x + s];
    __syncthreads();
  }
  if (threadIdx.x == 0)
    masks[b*NBv + j] = (sigmoidf_(red[0] + rb[j]) > SKIPT) ? 1.f : 0.f;
}
// layernorm, writes fp32 + bf16 copies
__global__ __launch_bounds__(256)
void ln_kernel(const float* __restrict__ x, const float* __restrict__ g,
               const float* __restrict__ bta, float* __restrict__ y, bf16* __restrict__ yb){
  __shared__ float r1[256], r2[256];
  int row = blockIdx.x;
  const float* xr = x + (size_t)row*Dd;
  float s = 0.f, s2 = 0.f;
  for (int d = threadIdx.x; d < Dd; d += 256){ float v = xr[d]; s += v; s2 += v*v; }
  r1[threadIdx.x] = s; r2[threadIdx.x] = s2; __syncthreads();
  for (int st = 128; st > 0; st >>= 1){
    if (threadIdx.x < st){ r1[threadIdx.x] += r1[threadIdx.x+st]; r2[threadIdx.x] += r2[threadIdx.x+st]; }
    __syncthreads();
  }
  float mean = r1[0]*(1.f/Dd);
  float var  = r2[0]*(1.f/Dd) - mean*mean;
  float inv  = rsqrtf(var + 1e-6f);
  float* yr = y + (size_t)row*Dd;
  bf16*  br = yb + (size_t)row*Dd;
  for (int d = threadIdx.x; d < Dd; d += 256){
    float v = (xr[d]-mean)*inv*g[d] + bta[d];
    yr[d] = v; br[d] = (bf16)v;
  }
}
__global__ void resadd_kernel(float* __restrict__ out, const float* __restrict__ src,
                              const float* __restrict__ masks, int blk){
  size_t i = (size_t)blockIdx.x*256 + threadIdx.x;
  int b = (int)(i / ((size_t)Ss*Dd));
  out[i] += masks[b*NBv + blk] * src[i];
}
// mamba: causal depthwise conv (k=4) + silu; fused dt = softplus(xc*wdt+dtb)
__global__ void convdt_kernel(const float* __restrict__ xz, const float* __restrict__ cw,
                              const float* __restrict__ wdt, const float* __restrict__ dtb,
                              float* __restrict__ xc, bf16* __restrict__ xcb,
                              float* __restrict__ dt){
  size_t i = (size_t)blockIdx.x*256 + threadIdx.x;
  if (i >= (size_t)TOK*DI) return;
  int c = (int)(i % DI);
  int s = (int)((i / DI) % Ss);
  int b = (int)(i / ((size_t)Ss*DI));
  float acc = 0.f;
  #pragma unroll
  for (int t = 0; t < 4; t++){
    int sp = s - 3 + t;
    if (sp >= 0) acc += cw[t*DI + c] * xz[((size_t)(b*Ss + sp))*(2*DI) + c];
  }
  float xcv = acc * sigmoidf_(acc);
  xc[i] = xcv; xcb[i] = (bf16)xcv;
  dt[i] = softplusf_(xcv * wdt[c] + dtb[c]);
}
// selective scan: lane-per-channel, 64 states in VGPRs; fused d_skip + silu(z); bf16 out
__global__ __launch_bounds__(256)
void scan_kernel(const float* __restrict__ xz, const float* __restrict__ xc,
                 const float* __restrict__ dt, const float* __restrict__ bc,
                 const float* __restrict__ alog, const float* __restrict__ dskip,
                 bf16* __restrict__ y){
  int g = blockIdx.x*256 + threadIdx.x;    // (b, c)
  int b = g / DI, c = g % DI;
  float A  = -__expf(alog[c]);
  float ds = dskip[c];
  float hst[64];
  #pragma unroll
  for (int n = 0; n < 64; n++) hst[n] = 0.f;
  const float* bcb = bc + (size_t)b * Ss * 128;
  for (int t = 0; t < Ss; t++){
    size_t tok = (size_t)b*Ss + t;
    float xcv = xc[tok*DI + c];
    float dtv = dt[tok*DI + c];
    float zv  = xz[tok*(2*DI) + DI + c];
    float decay = __expf(dtv * A);
    float u = dtv * xcv;
    const float* Bt = bcb + (size_t)t*128;
    const float* Ct = Bt + 64;
    float acc = 0.f;
    #pragma unroll
    for (int n = 0; n < 64; n++){
      hst[n] = decay*hst[n] + u*Bt[n];
      acc += hst[n]*Ct[n];
    }
    float sz = zv * sigmoidf_(zv);
    y[tok*DI + c] = (bf16)((acc + ds*xcv) * sz);
  }
}
__global__ void tc_combine_kernel(float* __restrict__ out, const float* __restrict__ xln,
                                  const float* __restrict__ tb, const float* __restrict__ theta,
                                  const float* __restrict__ masks){
  size_t i = (size_t)blockIdx.x*256 + threadIdx.x;
  int b = (int)(i / ((size_t)Ss*Dd));
  float th = theta[0];
  out[i] += masks[b*NBv + 1] * (__cosf(th)*xln[i] + __sinf(th)*tanhf(tb[i]));
}
__global__ void haar_kernel(const float* __restrict__ xln, bf16* __restrict__ a,
                            float* __restrict__ det){
  size_t i = (size_t)blockIdx.x*256 + threadIdx.x;
  if (i >= (size_t)TOKH*Dd) return;
  int d = (int)(i % Dd);
  int l = (int)((i / Dd) % Ll);
  int b = (int)(i / ((size_t)Ll*Dd));
  float xe = xln[((size_t)(b*Ss + 2*l    ))*Dd + d];
  float xo = xln[((size_t)(b*Ss + 2*l + 1))*Dd + d];
  a[i]   = (bf16)((xe + xo) * HAARI);
  det[i] = (xe - xo) * HAARI;
}
__global__ void invhaar_kernel(const float* __restrict__ o2, const float* __restrict__ det,
                               float* __restrict__ y){
  size_t i = (size_t)blockIdx.x*256 + threadIdx.x;
  if (i >= (size_t)TOKH*Dd) return;
  int d = (int)(i % Dd);
  int l = (int)((i / Dd) % Ll);
  int b = (int)(i / ((size_t)Ll*Dd));
  float o = o2[i], dt_ = det[i];
  y[((size_t)(b*Ss + 2*l    ))*Dd + d] = (o + dt_) * HAARI;
  y[((size_t)(b*Ss + 2*l + 1))*Dd + d] = (o - dt_) * HAARI;
}
__global__ void convres_kernel(float* __restrict__ out, const float* __restrict__ y,
                               const float* __restrict__ cw, const float* __restrict__ masks){
  size_t i = (size_t)blockIdx.x*256 + threadIdx.x;
  int d = (int)(i % Dd);
  int s = (int)((i / Dd) % Ss);
  int b = (int)(i / ((size_t)Ss*Dd));
  float acc = 0.f;
  #pragma unroll
  for (int t = 0; t < 4; t++){
    int sp = s - 2 + t;
    if (sp >= 0 && sp < Ss) acc += cw[t*Dd + d] * y[((size_t)(b*Ss + sp))*Dd + d];
  }
  out[i] += masks[b*NBv + 2] * acc;
}
__global__ void moe_logits_kernel(const float* __restrict__ xln, const float* __restrict__ wg,
                                  float* __restrict__ logits){
  int i = blockIdx.x*256 + threadIdx.x;
  if (i >= TOK*Ee) return;
  int t = i >> 3, e = i & 7;
  const float* xr = xln + (size_t)t*Dd;
  float acc = 0.f;
  for (int d = 0; d < Dd; d++) acc += xr[d] * wg[d*Ee + e];
  logits[i] = acc;
}
__global__ void moe_topk_kernel(const float* __restrict__ logits, float* __restrict__ gates){
  int t = blockIdx.x*256 + threadIdx.x;
  if (t >= TOK) return;
  float l[Ee];
  #pragma unroll
  for (int e = 0; e < Ee; e++) l[e] = logits[t*Ee + e];
  int i1 = 0;
  #pragma unroll
  for (int e = 1; e < Ee; e++) if (l[e] > l[i1]) i1 = e;
  int i2 = (i1 == 0) ? 1 : 0;
  #pragma unroll
  for (int e = 0; e < Ee; e++) if (e != i1 && l[e] > l[i2]) i2 = e;
  float mx = l[i1];
  float e1 = __expf(l[i1] - mx), e2 = __expf(l[i2] - mx);
  float inv = 1.f / (e1 + e2);
  #pragma unroll
  for (int e = 0; e < Ee; e++) gates[t*Ee + e] = 0.f;
  gates[t*Ee + i1] = e1 * inv;
  gates[t*Ee + i2] = e2 * inv;
}

} // anonymous namespace

// ================== driver ==================
extern "C" void kernel_launch(void* const* d_in, const int* in_sizes, int n_in,
                              void* d_out, int out_size, void* d_ws, size_t ws_size,
                              hipStream_t stream)
{
  (void)in_sizes; (void)n_in; (void)out_size; (void)ws_size;
  const float* h0    = (const float*)d_in[0];
  const float* rw    = (const float*)d_in[1];
  const float* rb    = (const float*)d_in[2];
  const float* lng   = (const float*)d_in[3];
  const float* lnb   = (const float*)d_in[4];
  const float* mwin  = (const float*)d_in[5];
  const float* mconv = (const float*)d_in[6];
  const float* malog = (const float*)d_in[7];
  const float* mwbc  = (const float*)d_in[8];
  const float* mwdt  = (const float*)d_in[9];
  const float* mdtb  = (const float*)d_in[10];
  const float* mdsk  = (const float*)d_in[11];
  const float* mwout = (const float*)d_in[12];
  const float* tcth  = (const float*)d_in[13];
  const float* tcw   = (const float*)d_in[14];
  const float* tcb   = (const float*)d_in[15];
  const float* wq    = (const float*)d_in[16];
  const float* wk    = (const float*)d_in[17];
  const float* wv    = (const float*)d_in[18];
  const float* wvo   = (const float*)d_in[19];
  const float* wconv = (const float*)d_in[20];
  const float* mwg   = (const float*)d_in[21];
  const float* mw1   = (const float*)d_in[22];
  const float* mb1   = (const float*)d_in[23];
  const float* mw2   = (const float*)d_in[24];
  const float* mb2   = (const float*)d_in[25];

  float* out = (float*)d_out;
  float* ws  = (float*)d_ws;
  float* masks  = ws + OFF_MASK;
  float* pooled = ws + OFF_POOL;
  float* xln    = ws + OFF_XLN;
  float* big    = ws + OFF_BIG;
  float* xcbuf  = ws + OFF_XC;
  float* dtbuf  = ws + OFF_DT;
  float* bcbuf  = ws + OFF_BC;
  float* ybuf   = ws + OFF_Y;
  float* blk    = ws + OFF_BLK;
  bf16*  bws    = (bf16*)(ws + OFF_BFB);
  bf16*  bf_xln = bws + BF_XLN;
  bf16*  bf_scr = bws + BF_SCR;

  const int TOT = TOK * Dd;
  const dim3 blk256(256);
  auto ew = [](size_t n){ return dim3((unsigned)((n + 255) / 256)); };

  copy_kernel<<<ew(TOT), blk256, 0, stream>>>(h0, out, TOT);
  pool_kernel<<<ew(Bb*Dd), blk256, 0, stream>>>(h0, pooled);
  router_kernel<<<dim3(Bb*NBv), blk256, 0, stream>>>(pooled, rw, rb, masks);

  // -------- branch 0: Mamba --------
  bf16* bf_xc = bf_scr;
  bf16* bf_y  = bf_scr + (size_t)TOK*DI;
  ln_kernel<<<dim3(TOK), blk256, 0, stream>>>(out, lng + 0*Dd, lnb + 0*Dd, xln, bf_xln);
  gemm_bf16<EPI_NONE,false,false><<<dim3(2*DI/BN, TOK/BM), blk256, 0, stream>>>(
      bf_xln, mwin, nullptr, nullptr, 0, big, TOK, 2*DI, Dd);
  convdt_kernel<<<ew((size_t)TOK*DI), blk256, 0, stream>>>(big, mconv, mwdt, mdtb,
                                                           xcbuf, bf_xc, dtbuf);
  gemm_bf16<EPI_NONE,false,false><<<dim3(128/BN, TOK/BM), blk256, 0, stream>>>(
      bf_xc, mwbc, nullptr, nullptr, 0, bcbuf, TOK, 128, DI);
  scan_kernel<<<dim3((Bb*DI)/256), blk256, 0, stream>>>(big, xcbuf, dtbuf, bcbuf,
                                                        malog, mdsk, bf_y);
  gemm_bf16<EPI_NONE,false,false><<<dim3(Dd/BN, TOK/BM), blk256, 0, stream>>>(
      bf_y, mwout, nullptr, nullptr, 0, blk, TOK, Dd, DI);
  resadd_kernel<<<ew(TOT), blk256, 0, stream>>>(out, blk, masks, 0);

  // -------- branch 1: TimeCrystal --------
  ln_kernel<<<dim3(TOK), blk256, 0, stream>>>(out, lng + 1*Dd, lnb + 1*Dd, xln, bf_xln);
  gemm_bf16<EPI_BIAS,false,false><<<dim3(Dd/BN, TOK/BM), blk256, 0, stream>>>(
      bf_xln, tcw, tcb, nullptr, 0, big, TOK, Dd, Dd);
  tc_combine_kernel<<<ew(TOT), blk256, 0, stream>>>(out, xln, big, tcth, masks);

  // -------- branch 2: WLAM --------
  ln_kernel<<<dim3(TOK), blk256, 0, stream>>>(out, lng + 2*Dd, lnb + 2*Dd, xln, bf_xln);
  bf16* bf_a  = bf_scr;
  bf16* bf_q  = bf_scr + 1*(size_t)TOKH*Dd;
  bf16* bf_k  = bf_scr + 2*(size_t)TOKH*Dd;
  bf16* bf_v  = bf_scr + 3*(size_t)TOKH*Dd;
  bf16* bf_oh = bf_scr + 4*(size_t)TOKH*Dd;
  float* det  = dtbuf;
  float* o2   = xcbuf;
  haar_kernel<<<ew((size_t)TOKH*Dd), blk256, 0, stream>>>(xln, bf_a, det);
  gemm_bf16<EPI_NONE,false,true><<<dim3(Dd/BN, TOKH/BM), blk256, 0, stream>>>(
      bf_a, wq, nullptr, nullptr, 0, bf_q, TOKH, Dd, Dd);
  gemm_bf16<EPI_NONE,false,true><<<dim3(Dd/BN, TOKH/BM), blk256, 0, stream>>>(
      bf_a, wk, nullptr, nullptr, 0, bf_k, TOKH, Dd, Dd);
  gemm_bf16<EPI_NONE,false,true><<<dim3(Dd/BN, TOKH/BM), blk256, 0, stream>>>(
      bf_a, wv, nullptr, nullptr, 0, bf_v, TOKH, Dd, Dd);
  attn_kernel<<<dim3(Ll/128, Bb*NH), blk256, 0, stream>>>(bf_q, bf_k, bf_v, bf_oh);
  gemm_bf16<EPI_NONE,false,false><<<dim3(Dd/BN, TOKH/BM), blk256, 0, stream>>>(
      bf_oh, wvo, nullptr, nullptr, 0, o2, TOKH, Dd, Dd);
  invhaar_kernel<<<ew((size_t)TOKH*Dd), blk256, 0, stream>>>(o2, det, ybuf);
  convres_kernel<<<ew(TOT), blk256, 0, stream>>>(out, ybuf, wconv, masks);

  // -------- branch 3: MoE (top-2, block-skip on zero gates) --------
  ln_kernel<<<dim3(TOK), blk256, 0, stream>>>(out, lng + 3*Dd, lnb + 3*Dd, xln, bf_xln);
  float* logits = bcbuf;
  float* gates  = bcbuf + (size_t)TOK*Ee;
  bf16*  bf_hff = bf_scr;
  moe_logits_kernel<<<ew(TOK*Ee), blk256, 0, stream>>>(xln, mwg, logits);
  moe_topk_kernel<<<ew(TOK), blk256, 0, stream>>>(logits, gates);
  zero_kernel<<<ew(TOT), blk256, 0, stream>>>(blk, TOT);
  for (int e = 0; e < Ee; e++) {
    const float* w1e = mw1 + (size_t)e*Dd*FF;
    const float* b1e = mb1 + (size_t)e*FF;
    const float* w2e = mw2 + (size_t)e*FF*Dd;
    const float* b2e = mb2 + (size_t)e*Dd;
    gemm_bf16<EPI_GELU,true,true><<<dim3(FF/BN, TOK/BM), blk256, 0, stream>>>(
        bf_xln, w1e, b1e, gates + e, Ee, bf_hff, TOK, FF, Dd);
    gemm_bf16<EPI_GATE_ACC,true,false><<<dim3(Dd/BN, TOK/BM), blk256, 0, stream>>>(
        bf_hff, w2e, b2e, gates + e, Ee, blk, TOK, Dd, FF);
  }
  resadd_kernel<<<ew(TOT), blk256, 0, stream>>>(out, blk, masks, 3);
}
